// TextGenerationModel_60224031425307
// MI455X (gfx1250) — matrix operand order repaired and checked by reading, not compiled, not run on hardware
//
#include <hip/hip_runtime.h>

// 2-layer LSTM text-gen, persistent WMMA kernel for gfx1250 (MI455X).
// bf16 WMMA (v_wmma_f32_16x16x32_bf16) with f32 accumulators; weights
// pre-transposed to [N][K] bf16 so B-fragments are two contiguous 8-half runs
// per lane. A-tiles staged into LDS via the Tensor Data Mover
// (tensor_load_to_lds + s_wait_tensorcnt). Each wave owns 2 M-tiles x
// 1 N-tile x all 4 gates, so B-fragments are reused and the LSTM cell
// update (c,h) happens entirely in registers. K loops are unrolled x2
// only, to keep VGPR pressure below the spill threshold.

#define LAYERS 2
#define VOCAB  256
#define HID    1024
#define BATCH  128
#define SEQT   512
#define NWG    32
#define TPB    256

typedef __attribute__((ext_vector_type(16))) __bf16 v16bf;
typedef __attribute__((ext_vector_type(8)))  float  v8f;
typedef __attribute__((ext_vector_type(4)))  unsigned int u32x4;
typedef __attribute__((ext_vector_type(8)))  int i32x8;
typedef __attribute__((ext_vector_type(4)))  int i32x4;

struct __align__(16) U4 { unsigned x, y, z, w; };
union Frag { v16bf v; U4 q[2]; };

__device__ __forceinline__ unsigned short f32_to_bf16(float f) {
  unsigned u = __float_as_uint(f);
  u += 0x7FFFu + ((u >> 16) & 1u);      // round-to-nearest-even
  return (unsigned short)(u >> 16);
}

__device__ __forceinline__ float sigm(float x) {
  return 1.0f / (1.0f + __expf(-x));
}

// 2D tile load, Global -> LDS, via the Tensor Data Mover.
// rowU8 / strideU8 are in 8-byte units (data_size code 3).
__device__ __forceinline__ void tdm_load_2d(unsigned lds_addr, const void* gptr,
                                            unsigned rowU8, unsigned rows,
                                            unsigned strideU8) {
  unsigned long long ga = (unsigned long long)(size_t)gptr;
  u32x4 g0 = { 1u,                                  // count=1 valid user D#
               lds_addr,                            // lds_addr [63:32]
               (unsigned)(ga & 0xFFFFFFFFu),        // global_addr lo
               (unsigned)((ga >> 32) & 0x1FFFFFFu) | 0x80000000u }; // hi + type=2
  i32x8 g1 = { (int)(3u << 16),        // data_size = 8B; no multicast/pad/iter
               (int)(rowU8 << 16),     // tensor_dim0[15:0]
               (int)(rows  << 16),     // tensor_dim0 hi=0 ; tensor_dim1[15:0]
               (int)(rowU8 << 16),     // tensor_dim1 hi=0 ; tile_dim0
               (int)rows,              // tile_dim1 ; tile_dim2=0
               (int)strideU8,          // tensor_dim0_stride lo
               0, 0 };
  i32x4 z = {0, 0, 0, 0};
#if defined(__clang_major__) && (__clang_major__ >= 23)
  i32x8 z8 = {0, 0, 0, 0, 0, 0, 0, 0};
  __builtin_amdgcn_tensor_load_to_lds(g0, g1, z, z, z8, 0);
#else
  __builtin_amdgcn_tensor_load_to_lds(g0, g1, z, z, 0);
#endif
}

__global__ void transpose_to_bf16(const float* __restrict__ src,    // [K][N]
                                  unsigned short* __restrict__ dst, // [N][K]
                                  int K, int N) {
  int total = K * N;
  for (int idx = blockIdx.x * blockDim.x + threadIdx.x; idx < total;
       idx += gridDim.x * blockDim.x) {
    int n = idx / K;
    int k = idx - n * K;
    dst[idx] = f32_to_bf16(src[(size_t)k * N + n]);
  }
}

__device__ __forceinline__ void grid_sync(unsigned* cnt, unsigned& epoch) {
  __syncthreads();
  if (threadIdx.x == 0) {
    __threadfence();
    atomicAdd(cnt, 1u);
    epoch += NWG;
    while (*((volatile unsigned*)cnt) < epoch) {
      __builtin_amdgcn_s_sleep(2);
    }
  }
  __syncthreads();
}

__global__ __launch_bounds__(TPB)
void lstm_persistent(const int* __restrict__ xtok,               // [B,T]
                     const float* __restrict__ Wgx0, const float* __restrict__ Wix0,
                     const float* __restrict__ Wfx0, const float* __restrict__ Wox0,
                     const float* __restrict__ bg,  const float* __restrict__ bi,
                     const float* __restrict__ bfv, const float* __restrict__ bo,
                     const float* __restrict__ bp,
                     const unsigned short* __restrict__ WhT,     // [L,4,H][H]  bf16 N-major
                     const unsigned short* __restrict__ WxT,     // [L,4,H][V]
                     const unsigned short* __restrict__ WpT,     // [L,V][H]
                     unsigned short* __restrict__ hbf,           // [B,H] bf16 state
                     float* __restrict__ cst,                    // [B,H] f32 cell
                     unsigned short* __restrict__ pseq,          // [T,B,V] bf16 (layer0 out)
                     float* __restrict__ out,                    // [T,B,V] f32
                     unsigned* __restrict__ cnt)
{
  __shared__ unsigned short Ash[32 * HID];    // 64 KB: 32 h-rows for this block
  __shared__ unsigned short Psh[32 * VOCAB];  // 16 KB: 32 p-rows (layer-1 input)

  const int tid     = threadIdx.x;
  const int lane    = tid & 31;
  const int wg      = (blockIdx.x * TPB + tid) >> 5;   // 0..255
  const int halfSel = lane >> 4;                       // 0 or 1
  const int lane16  = lane & 15;
  const int aOff    = halfSel * 8;                     // A-frag K sub-offset
  const int bOff    = halfSel * 8;                     // B-frag K sub-offset

  const int mtg   = wg >> 6;                 // 32-row batch group 0..3 (block-uniform)
  const int nt    = wg & 63;                 // H tile 0..63
  const int nOut  = nt * 16 + lane16;        // output column within gate
  const int mRow0 = mtg * 32 + halfSel * 8;  // C-frag row base, M-tile 0
  const int aRow0 = lane16 * HID + aOff;     // A-frag LDS offsets (halves)
  const int aRow1 = (16 + lane16) * HID + aOff;

  unsigned epoch = 0;

  for (int l = 0; l < LAYERS; ++l) {
    // zero h, c for this layer
    for (int i = blockIdx.x * TPB + tid; i < BATCH * HID; i += NWG * TPB) {
      hbf[i] = 0;
      cst[i] = 0.0f;
    }
    grid_sync(cnt, epoch);

    for (int t = 0; t < SEQT; ++t) {
      // ---- stage A tiles via TDM (wave 0 issues; TDM ignores EXEC) ----
      if (tid < 32) {
        tdm_load_2d((unsigned)(size_t)Ash,
                    hbf + (size_t)mtg * 32 * HID,
                    (HID * 2) / 8, 32, (HID * 2) / 8);
        if (l == 1)
          tdm_load_2d((unsigned)(size_t)Psh,
                      pseq + ((size_t)t * BATCH + mtg * 32) * VOCAB,
                      (VOCAB * 2) / 8, 32, (VOCAB * 2) / 8);
        __builtin_amdgcn_s_wait_tensorcnt(0);
      }
      __syncthreads();

      // ---- gate phase: 2 M-tiles x 1 N-tile, all 4 gates in registers ----
      v8f acc[4][2];
      #pragma unroll
      for (int g = 0; g < 4; ++g)
        #pragma unroll
        for (int mm = 0; mm < 2; ++mm)
          #pragma unroll
          for (int r = 0; r < 8; ++r) acc[g][mm][r] = 0.0f;

      // recurrent part: h[B,H] @ Wh[H,H], bf16 WMMA, K = HID
      #pragma unroll 2
      for (int kb = 0; kb < HID; kb += 32) {
        Frag a0, a1;
        const unsigned short* ap0 = Ash + aRow0 + kb;
        const unsigned short* ap1 = Ash + aRow1 + kb;
        a0.q[0] = *(const U4*)(ap0);
        a0.q[1] = *(const U4*)(ap0 + 16);
        a1.q[0] = *(const U4*)(ap1);
        a1.q[1] = *(const U4*)(ap1 + 16);
        #pragma unroll
        for (int g = 0; g < 4; ++g) {
          Frag b;
          const unsigned short* bptr =
              WhT + ((size_t)(l * 4 + g) * HID + nOut) * HID + kb + bOff;
          b.q[0] = *(const U4*)(bptr);
          b.q[1] = *(const U4*)(bptr + 16);
          if (g == 0) __builtin_prefetch(bptr + 64);
          acc[g][0] = __builtin_amdgcn_wmma_f32_16x16x32_bf16(
              false, a0.v, false, b.v, (short)0, acc[g][0], false, false);
          acc[g][1] = __builtin_amdgcn_wmma_f32_16x16x32_bf16(
              false, a1.v, false, b.v, (short)0, acc[g][1], false, false);
        }
      }

      if (l == 0) {
        // one-hot input path: gather rows of Wx (f32) by token id
        #pragma unroll
        for (int mm = 0; mm < 2; ++mm)
          #pragma unroll
          for (int r = 0; r < 8; ++r) {
            const int m   = mRow0 + mm * 16 + r;
            const int tok = xtok[(size_t)m * SEQT + t];
            const size_t o = (size_t)tok * HID + nOut;
            acc[0][mm][r] += Wgx0[o];
            acc[1][mm][r] += Wix0[o];
            acc[2][mm][r] += Wfx0[o];
            acc[3][mm][r] += Wox0[o];
          }
      } else {
        // dense input path: p0[t][B,V] @ Wx[V,H], bf16 WMMA, K = VOCAB
        #pragma unroll 2
        for (int kb = 0; kb < VOCAB; kb += 32) {
          Frag a0, a1;
          const unsigned short* ap0 = Psh + lane16 * VOCAB + kb + aOff;
          const unsigned short* ap1 = Psh + (16 + lane16) * VOCAB + kb + aOff;
          a0.q[0] = *(const U4*)(ap0);
          a0.q[1] = *(const U4*)(ap0 + 16);
          a1.q[0] = *(const U4*)(ap1);
          a1.q[1] = *(const U4*)(ap1 + 16);
          #pragma unroll
          for (int g = 0; g < 4; ++g) {
            Frag b;
            const unsigned short* bptr =
                WxT + ((size_t)(l * 4 + g) * HID + nOut) * VOCAB + kb + bOff;
            b.q[0] = *(const U4*)(bptr);
            b.q[1] = *(const U4*)(bptr + 16);
            acc[g][0] = __builtin_amdgcn_wmma_f32_16x16x32_bf16(
                false, a0.v, false, b.v, (short)0, acc[g][0], false, false);
            acc[g][1] = __builtin_amdgcn_wmma_f32_16x16x32_bf16(
                false, a1.v, false, b.v, (short)0, acc[g][1], false, false);
          }
        }
      }

      // activations + cell/state update (C-frag: m = mRow0 + mm*16 + r, n = nOut)
      {
        const float bgv = bg [l * HID + nOut];
        const float biv = bi [l * HID + nOut];
        const float bfx = bfv[l * HID + nOut];
        const float bov = bo [l * HID + nOut];
        #pragma unroll
        for (int mm = 0; mm < 2; ++mm)
          #pragma unroll
          for (int r = 0; r < 8; ++r) {
            const int m = mRow0 + mm * 16 + r;
            const size_t idx = (size_t)m * HID + nOut;
            const float gv = tanhf(acc[0][mm][r] + bgv);
            const float iv = sigm (acc[1][mm][r] + biv);
            const float fv = sigm (acc[2][mm][r] + bfx);
            const float ov = sigm (acc[3][mm][r] + bov);
            const float cn = gv * iv + cst[idx] * fv;
            cst[idx] = cn;
            hbf[idx] = f32_to_bf16(tanhf(cn) * ov);
          }
      }
      grid_sync(cnt, epoch);

      // ---------- projection phase: p = h @ Wp + bp (blocks 0..15) ----------
      if (blockIdx.x < 16) {
        const int pmt  = blockIdx.x >> 1;     // batch tile, block-uniform
        const int pnt  = wg & 15;             // vocab tile 0..15
        const int pCol = pnt * 16 + lane16;
        const int pmB  = pmt * 16 + halfSel * 8;
        if (tid < 32) {
          tdm_load_2d((unsigned)(size_t)Ash,
                      hbf + (size_t)pmt * 16 * HID,
                      (HID * 2) / 8, 16, (HID * 2) / 8);
          __builtin_amdgcn_s_wait_tensorcnt(0);
        }
        __syncthreads();

        v8f pacc;
        const float bpv = bp[l * VOCAB + pCol];
        #pragma unroll
        for (int r = 0; r < 8; ++r) pacc[r] = bpv;

        #pragma unroll 2
        for (int kb = 0; kb < HID; kb += 32) {
          Frag a;
          const unsigned short* ap = Ash + aRow0 + kb;
          a.q[0] = *(const U4*)(ap);
          a.q[1] = *(const U4*)(ap + 16);
          Frag b;
          const unsigned short* bptr =
              WpT + ((size_t)l * VOCAB + pCol) * HID + kb + bOff;
          b.q[0] = *(const U4*)(bptr);
          b.q[1] = *(const U4*)(bptr + 16);
          pacc = __builtin_amdgcn_wmma_f32_16x16x32_bf16(
              false, a.v, false, b.v, (short)0, pacc, false, false);
        }
        if (l == 0) {
          #pragma unroll
          for (int r = 0; r < 8; ++r)
            pseq[((size_t)t * BATCH + pmB + r) * VOCAB + pCol] = f32_to_bf16(pacc[r]);
        } else {
          #pragma unroll
          for (int r = 0; r < 8; ++r)
            out[((size_t)t * BATCH + pmB + r) * VOCAB + pCol] = pacc[r];
        }
      }
      grid_sync(cnt, epoch);
    }
  }
}

extern "C" void kernel_launch(void* const* d_in, const int* in_sizes, int n_in,
                              void* d_out, int out_size, void* d_ws, size_t ws_size,
                              hipStream_t stream) {
  (void)in_sizes; (void)n_in; (void)out_size; (void)ws_size;

  const int*   x    = (const int*)  d_in[0];
  const float* W_gx = (const float*)d_in[1];
  const float* W_gh = (const float*)d_in[2];
  const float* W_ix = (const float*)d_in[3];
  const float* W_ih = (const float*)d_in[4];
  const float* W_fx = (const float*)d_in[5];
  const float* W_fh = (const float*)d_in[6];
  const float* W_ox = (const float*)d_in[7];
  const float* W_oh = (const float*)d_in[8];
  const float* W_ph = (const float*)d_in[9];
  const float* b_g  = (const float*)d_in[10];
  const float* b_i  = (const float*)d_in[11];
  const float* b_f  = (const float*)d_in[12];
  const float* b_o  = (const float*)d_in[13];
  const float* b_p  = (const float*)d_in[14];

  char* ws = (char*)d_ws;
  size_t off = 0;
  unsigned* cnt        = (unsigned*)(ws + off);       off += 256;
  unsigned short* WhT  = (unsigned short*)(ws + off); off += (size_t)LAYERS * 4 * HID * HID * 2;
  unsigned short* WxT  = (unsigned short*)(ws + off); off += (size_t)LAYERS * 4 * HID * VOCAB * 2;
  unsigned short* WpT  = (unsigned short*)(ws + off); off += (size_t)LAYERS * VOCAB * HID * 2;
  unsigned short* hbf  = (unsigned short*)(ws + off); off += (size_t)BATCH * HID * 2;
  float* cst           = (float*)(ws + off);          off += (size_t)BATCH * HID * 4;
  unsigned short* pseq = (unsigned short*)(ws + off); off += (size_t)SEQT * BATCH * VOCAB * 2;

  hipMemsetAsync(cnt, 0, 256, stream);   // graph-capture-safe barrier reset

  const float* WhSrc[4] = {W_gh, W_ih, W_fh, W_oh};
  const float* WxSrc[4] = {W_gx, W_ix, W_fx, W_ox};
  for (int l = 0; l < LAYERS; ++l) {
    for (int g = 0; g < 4; ++g) {
      int totH = HID * HID;
      transpose_to_bf16<<<(totH + 255) / 256, 256, 0, stream>>>(
          WhSrc[g] + (size_t)l * HID * HID,
          WhT + (size_t)(l * 4 + g) * HID * HID, HID, HID);
      int totX = HID * VOCAB;
      transpose_to_bf16<<<(totX + 255) / 256, 256, 0, stream>>>(
          WxSrc[g] + (size_t)l * VOCAB * HID,
          WxT + (size_t)(l * 4 + g) * HID * VOCAB, VOCAB, HID);
    }
    int totP = VOCAB * HID;
    transpose_to_bf16<<<(totP + 255) / 256, 256, 0, stream>>>(
        W_ph + (size_t)l * HID * VOCAB,
        WpT + (size_t)l * VOCAB * HID, HID, VOCAB);
  }

  lstm_persistent<<<NWG, TPB, 0, stream>>>(
      x, W_gx, W_ix, W_fx, W_ox,
      b_g, b_i, b_f, b_o, b_p,
      WhT, WxT, WpT, hbf, cst, pseq,
      (float*)d_out, cnt);
}
